// Sequence_1623497637992
// MI455X (gfx1250) — matrix operand order repaired and checked by reading, not compiled, not run on hardware
//
#include <hip/hip_runtime.h>

#define HIDDEN 51
#define NG 204          // 4*HIDDEN
#define T_LEN 8192
#define OUT_CH 2
#define NTILE 14        // 13 gate tiles + 1 output tile
#define GSTRIDE 20      // gates_lds row stride (floats), 80B -> 16B aligned float4
#define HSTRIDE 36      // h_lds row stride (uints), 144B -> 16B aligned, conflict-free

typedef __attribute__((ext_vector_type(16))) __bf16    v16bf;
typedef __attribute__((ext_vector_type(8)))  float     v8f;
typedef __attribute__((ext_vector_type(8)))  unsigned  v8u;

__device__ inline unsigned short f2bf(float f) {
    unsigned u = __builtin_bit_cast(unsigned, f);
    u += 0x7FFFu + ((u >> 16) & 1u);          // round-to-nearest-even
    return (unsigned short)(u >> 16);
}
__device__ inline unsigned pack2(float a, float b) {
    return (unsigned)f2bf(a) | ((unsigned)f2bf(b) << 16);
}
// Opaque VGPR pins: stop LLVM from rematerializing loop-invariant values inside the time loop.
__device__ inline float pin_f32(float x) {
    float r; asm volatile("v_mov_b32 %0, %1" : "=v"(r) : "v"(x)); return r;
}
__device__ inline unsigned pin_u32(unsigned x) {
    unsigned r; asm volatile("v_mov_b32 %0, %1" : "=v"(r) : "v"(x)); return r;
}
__device__ inline float sigm(float x) {
    x = fminf(fmaxf(x, -30.f), 30.f);
    return 1.0f / (1.0f + __expf(-x));
}
__device__ inline float tanh_fast(float x) {
    x = fminf(fmaxf(x, -15.f), 15.f);
    float e = __expf(-2.0f * x);
    return (1.0f - e) / (1.0f + e);
}

// Load one 16x32 bf16 A fragment (WMMA A layout) from an h_lds row of packed pairs.
__device__ inline v16bf load_afrag(const unsigned* h_lds, unsigned hb, int c) {
    const unsigned base0 = hb + 16 * c;          // hb = lo*HSTRIDE + hi*8 (pinned)
    uint4 a = *(const uint4*)&h_lds[base0];      // VGPRs 0..3
    uint4 b = *(const uint4*)&h_lds[base0 + 4];  // VGPRs 4..7
    v8u u;
    u[0] = a.x; u[1] = a.y; u[2] = a.z; u[3] = a.w;
    u[4] = b.x; u[5] = b.y; u[6] = b.z; u[7] = b.w;
    return __builtin_bit_cast(v16bf, u);
}

__global__ __launch_bounds__(256, 1)
void lstm_wmma_kernel(const float* __restrict__ input,
                      const float* __restrict__ W_ih,
                      const float* __restrict__ W_hh,
                      const float* __restrict__ b_ih,
                      const float* __restrict__ b_hh,
                      const float* __restrict__ W_lin,
                      const float* __restrict__ b_lin,
                      float* __restrict__ out)
{
    // gates_lds[n*GSTRIDE + m]: n = global gate col (0..207), m = batch row in tile
    __shared__ alignas(16) float    gates_lds[208 * GSTRIDE];
    // h_lds[m*HSTRIDE + p]: packed bf16 pairs; p=0..25 -> h[2p,2p+1], p=26 -> (x_t,0), 27..31 zero
    __shared__ alignas(16) unsigned h_lds[16 * HSTRIDE];

    const int tid   = threadIdx.x;
    const int w     = __builtin_amdgcn_readfirstlane(tid >> 5);   // wave id as SGPR (uniform)
    const int lane  = tid & 31;
    const int lo    = lane & 15;
    const int hi    = lane >> 4;
    const int bbase = blockIdx.x * 16;

    // ---------------- Build resident B fragments + bias (once) ----------------
    // B (32x16 bf16 per K-chunk): lane n=lo; K = 32*c + 16*hi + 2*v + {0,1}
    v16bf bf[2][2];
    float bias_r[2];
    #pragma unroll
    for (int q = 0; q < 2; ++q) {
        const int nt = w * 2 + q;
        bias_r[q] = 0.f;
        #pragma unroll
        for (int c = 0; c < 2; ++c) {
            v8u held;
            #pragma unroll
            for (int v = 0; v < 8; ++v) {
                const int k0 = 32 * c + 16 * hi + 2 * v;
                const int k1 = k0 + 1;
                float e0 = 0.f, e1 = 0.f;
                if (nt < 13) {                       // gate tiles: B[k][n] = W_hh[n][k]; row 52 = W_ih
                    const int n = nt * 16 + lo;
                    if (n < NG) {
                        if (k0 < HIDDEN)      e0 = W_hh[n * HIDDEN + k0];
                        else if (k0 == 52)    e0 = W_ih[n];
                        if (k1 < HIDDEN)      e1 = W_hh[n * HIDDEN + k1];
                    }
                } else if (nt == 13) {               // output tile: B[k][n] = W_lin[n][k]
                    const int n = lo;
                    if (n < OUT_CH) {
                        if (k0 < HIDDEN) e0 = W_lin[n * HIDDEN + k0];
                        if (k1 < HIDDEN) e1 = W_lin[n * HIDDEN + k1];
                    }
                }
                held[v] = pin_u32(pack2(e0, e1));    // pin packed bf16 pair in a VGPR
            }
            bf[q][c] = __builtin_bit_cast(v16bf, held);
        }
        if (nt < 13) {
            const int n = nt * 16 + lo;
            if (n < NG) bias_r[q] = b_ih[n] + b_hh[n];
        } else if (nt == 13 && lo < OUT_CH) {
            bias_r[q] = b_lin[lo];
        }
    }
    // Pin bias splat element-wise: 8 distinct opaque VGPRs -> resident C operand, no per-step copies.
    v8f acc_init[2];
    #pragma unroll
    for (int q = 0; q < 2; ++q) {
        #pragma unroll
        for (int r = 0; r < 8; ++r) acc_init[q][r] = pin_f32(bias_r[q]);
    }

    // ---------------- Hoisted, pinned loop-invariant addresses ----------------
    const unsigned hb = pin_u32((unsigned)(lo * HSTRIDE + hi * 8));          // A-frag load base
    unsigned gs[2];                                                          // gate-store bases
    #pragma unroll
    for (int q = 0; q < 2; ++q) {
        const int nt = w * 2 + q;
        const int n  = (nt < 13 ? nt : 0) * 16 + lo;
        gs[q] = pin_u32((unsigned)(n * GSTRIDE + 8 * hi));
    }
    unsigned gb[2], hs[2];                                                   // activation bases
    #pragma unroll
    for (int i = 0; i < 2; ++i) {
        const int jj  = w * 2 + i;
        const int idx = lane + 32 * (jj < 13 ? jj : 0);
        const int m   = idx & 15;
        const int p   = idx >> 4;
        gb[i] = pin_u32((unsigned)((2 * p) * GSTRIDE + m));
        hs[i] = pin_u32((unsigned)(m * HSTRIDE + p));
    }
    // Output store base (wave 6, lanes lo<2): rows m = 8*hi + r, out[b][lo][t]
    float* optr = out + ((size_t)(bbase + 8 * hi) * OUT_CH + lo) * T_LEN;
    // Input staging pointer (wave 7): row = bbase + lo
    const float* xptr = input + (size_t)(bbase + lo) * T_LEN;

    // ---------------- Init state ----------------
    for (int i = tid; i < 16 * HSTRIDE; i += 256) h_lds[i] = 0u;
    float2 cst[2] = { make_float2(0.f, 0.f), make_float2(0.f, 0.f) };
    __syncthreads();
    float xpend = 0.f;                               // x[t+1], staged one step ahead
    if (w == 7) {
        if (lane < 16)
            h_lds[lo * HSTRIDE + 26] = (unsigned)f2bf(xptr[0]);   // x[0] into K-slot 52 (pair 26)
        xpend = xptr[1 < T_LEN ? 1 : 0];
    }
    __syncthreads();

    // ---------------- Time loop ----------------
    #pragma unroll 1
    for (int t = 0; t < T_LEN; ++t) {
        float xfut = 0.f;
        if (w == 7) {                                // Phase A slot: prefetch x[t+2] (latency hidden)
            int ti = t + 2; ti = ti < T_LEN ? ti : T_LEN - 1;
            xfut = xptr[ti];
        } else {
            // Phase A: A fragments (h_{t-1} | x_t), then 2 WMMAs per owned tile
            v16bf a0 = load_afrag(h_lds, hb, 0);
            v16bf a1 = load_afrag(h_lds, hb, 1);
            #pragma unroll
            for (int q = 0; q < 2; ++q) {
                const int nt = w * 2 + q;
                v8f acc;
                acc = __builtin_amdgcn_wmma_f32_16x16x32_bf16(false, a0, false, bf[q][0],
                                                              (short)0, acc_init[q], false, false);
                acc = __builtin_amdgcn_wmma_f32_16x16x32_bf16(false, a1, false, bf[q][1],
                                                              (short)0, acc, false, false);
                if (nt < 13) {
                    *(float4*)&gates_lds[gs[q]]     = make_float4(acc[0], acc[1], acc[2], acc[3]);
                    *(float4*)&gates_lds[gs[q] + 4] = make_float4(acc[4], acc[5], acc[6], acc[7]);
                } else if (t > 0) {                  // out[t-1] = h_{t-1} @ W_lin^T + b_lin
                    if (lo < OUT_CH) {
                        float* op = optr + (t - 1);
                        #pragma unroll
                        for (int r = 0; r < 8; ++r)
                            op[r * (OUT_CH * T_LEN)] = acc[r];
                    }
                }
            }
        }
        __syncthreads();

        // Phase B: gate nonlinearities + state update (split by hidden-pair across waves)
        #pragma unroll
        for (int i = 0; i < 2; ++i) {
            const int jj = w * 2 + i;
            if (jj < 13) {                           // scalar branch
                const unsigned g = gb[i];
                float ig0 = gates_lds[g];
                float ig1 = gates_lds[g +       GSTRIDE];
                float fg0 = gates_lds[g +  51 * GSTRIDE];
                float fg1 = gates_lds[g +  52 * GSTRIDE];
                float gg0 = gates_lds[g + 102 * GSTRIDE];
                float gg1 = gates_lds[g + 103 * GSTRIDE];
                float og0 = gates_lds[g + 153 * GSTRIDE];
                float og1 = gates_lds[g + 154 * GSTRIDE];
                float c0 = sigm(fg0) * cst[i].x + sigm(ig0) * tanh_fast(gg0);
                float h0 = sigm(og0) * tanh_fast(c0);
                float c1 = sigm(fg1) * cst[i].y + sigm(ig1) * tanh_fast(gg1);
                float h1 = sigm(og1) * tanh_fast(c1);
                const int p = (int)(hs[i] % HSTRIDE);          // pair index (cheap? no -> see below)
                // pad column j=51 lives in pair 25 handled via jj==12 path:
                if (jj == 12 && lane >= 16) { c1 = 0.f; h1 = 0.f; }
                cst[i].x = c0; cst[i].y = c1;
                h_lds[hs[i]] = pack2(h0, h1);
                (void)p;
            }
        }
        if (w == 7) {                                // stage x[t+1] (value prefetched last step)
            if (lane < 16 && (t + 1) < T_LEN)
                h_lds[lo * HSTRIDE + 26] = (unsigned)f2bf(xpend);
            xpend = xfut;
        }
        __syncthreads();
    }

    // ---------------- Epilogue: out[T-1] = h_{T-1} @ W_lin^T + b_lin ----------------
    if (w == 6) {                                    // wave 6 owns the output tile (q=1)
        v16bf a0 = load_afrag(h_lds, hb, 0);
        v16bf a1 = load_afrag(h_lds, hb, 1);
        v8f acc;
        acc = __builtin_amdgcn_wmma_f32_16x16x32_bf16(false, a0, false, bf[1][0],
                                                      (short)0, acc_init[1], false, false);
        acc = __builtin_amdgcn_wmma_f32_16x16x32_bf16(false, a1, false, bf[1][1],
                                                      (short)0, acc, false, false);
        if (lo < OUT_CH) {
            float* op = optr + (T_LEN - 1);
            #pragma unroll
            for (int r = 0; r < 8; ++r)
                op[r * (OUT_CH * T_LEN)] = acc[r];
        }
    }
}

extern "C" void kernel_launch(void* const* d_in, const int* in_sizes, int n_in,
                              void* d_out, int out_size, void* d_ws, size_t ws_size,
                              hipStream_t stream) {
    const float* input = (const float*)d_in[0];
    const float* W_ih  = (const float*)d_in[1];
    const float* W_hh  = (const float*)d_in[2];
    const float* b_ih  = (const float*)d_in[3];
    const float* b_hh  = (const float*)d_in[4];
    const float* W_lin = (const float*)d_in[5];
    const float* b_lin = (const float*)d_in[6];
    float* out = (float*)d_out;

    // 64 batch tiles of 16 rows; 8 waves (256 threads) per WG split the 14 N-tiles.
    lstm_wmma_kernel<<<64, 256, 0, stream>>>(input, W_ih, W_hh, b_ih, b_hh, W_lin, b_lin, out);
}
